// BitLinear158Inference_38354057953285
// MI455X (gfx1250) — 8-bit matrix operand order repaired and checked by reading, compiled, not run on hardware
//
#include <hip/hip_runtime.h>

typedef __attribute__((ext_vector_type(8))) int v8i;

__global__ __launch_bounds__(256) void quant_act_kernel(
    const float* __restrict__ x, signed char* __restrict__ xq,
    float* __restrict__ gamma, int M) {
  const int K = 2048;
  const int wave = threadIdx.x >> 5;
  const int lane = threadIdx.x & 31;
  const int token = blockIdx.x * 8 + wave;
  if (token >= M) return;

  const float4* xp = (const float4*)(x + (size_t)token * K);
  float4 v[16];
  float m = 0.0f;
#pragma unroll
  for (int j = 0; j < 16; ++j) {
    v[j] = xp[lane + 32 * j];
    m = fmaxf(m, fmaxf(fmaxf(fabsf(v[j].x), fabsf(v[j].y)),
                       fmaxf(fabsf(v[j].z), fabsf(v[j].w))));
  }
#pragma unroll
  for (int off = 16; off > 0; off >>= 1)
    m = fmaxf(m, __shfl_xor(m, off, 32));
  m = fmaxf(m, 1e-5f);
  if (lane == 0) gamma[token] = m;

  const float s = 128.0f / m;
  unsigned int* qp = (unsigned int*)(xq + (size_t)token * K);
#pragma unroll
  for (int j = 0; j < 16; ++j) {
    int a = (int)fminf(fmaxf(rintf(v[j].x * s), -128.0f), 127.0f);
    int b = (int)fminf(fmaxf(rintf(v[j].y * s), -128.0f), 127.0f);
    int c = (int)fminf(fmaxf(rintf(v[j].z * s), -128.0f), 127.0f);
    int d = (int)fminf(fmaxf(rintf(v[j].w * s), -128.0f), 127.0f);
    unsigned int packed = ((unsigned)a & 255u) | (((unsigned)b & 255u) << 8) |
                          (((unsigned)c & 255u) << 16) |
                          (((unsigned)d & 255u) << 24);
    qp[lane + 32 * j] = packed;
  }
}

__global__ __launch_bounds__(256) void quant_wgt_kernel(
    const float* __restrict__ w, signed char* __restrict__ wq, int n4) {
  int i = blockIdx.x * 256 + threadIdx.x;
  if (i >= n4) return;
  float4 v = ((const float4*)w)[i];
  int a = (int)rintf(v.x);
  int b = (int)rintf(v.y);
  int c = (int)rintf(v.z);
  int d = (int)rintf(v.w);
  ((unsigned int*)wq)[i] = ((unsigned)a & 255u) | (((unsigned)b & 255u) << 8) |
                           (((unsigned)c & 255u) << 16) |
                           (((unsigned)d & 255u) << 24);
}

__global__ __launch_bounds__(256) void bitgemm_kernel(
    const signed char* __restrict__ A,
    const signed char* __restrict__ Bw,
    const float* __restrict__ gamma, const float* __restrict__ scaleP,
    const float* __restrict__ bias, float* __restrict__ out,
    int M, int N, int K) {
  const int lane = threadIdx.x & 31;
  const int wave = threadIdx.x >> 5;
  const int waveM = wave & 1;
  const int waveN = wave >> 1;
  const int mBase = blockIdx.x * 128 + waveM * 64;
  const int nBase = blockIdx.y * 128 + waveN * 32;
  const int r = lane & 15;
  const int hi = (lane >> 4) & 1;

  v8i acc[4][2];
#pragma unroll
  for (int i = 0; i < 4; ++i)
#pragma unroll
    for (int j = 0; j < 2; ++j)
      acc[i][j] = (v8i){0, 0, 0, 0, 0, 0, 0, 0};

  for (int k0 = 0; k0 < K; k0 += 64) {
    v8i aF[4];
#pragma unroll
    for (int i = 0; i < 4; ++i) {
      const signed char* p =
          A + (size_t)(mBase + 16 * i + r) * K + k0 + 8 * hi;
      unsigned long long d0 = *(const unsigned long long*)(p);
      unsigned long long d1 = *(const unsigned long long*)(p + 16);
      unsigned long long d2 = *(const unsigned long long*)(p + 32);
      unsigned long long d3 = *(const unsigned long long*)(p + 48);
      aF[i][0] = (int)(unsigned int)d0;
      aF[i][1] = (int)(unsigned int)(d0 >> 32);
      aF[i][2] = (int)(unsigned int)d1;
      aF[i][3] = (int)(unsigned int)(d1 >> 32);
      aF[i][4] = (int)(unsigned int)d2;
      aF[i][5] = (int)(unsigned int)(d2 >> 32);
      aF[i][6] = (int)(unsigned int)d3;
      aF[i][7] = (int)(unsigned int)(d3 >> 32);
    }
    v8i bF[2];
#pragma unroll
    for (int j = 0; j < 2; ++j) {
      const signed char* p =
          Bw + (size_t)(nBase + 16 * j + r) * K + k0 + 8 * hi;
      unsigned long long d0 = *(const unsigned long long*)(p);
      unsigned long long d1 = *(const unsigned long long*)(p + 16);
      unsigned long long d2 = *(const unsigned long long*)(p + 32);
      unsigned long long d3 = *(const unsigned long long*)(p + 48);
      bF[j] = (v8i){(int)(unsigned int)d0, (int)(unsigned int)(d0 >> 32),
                    (int)(unsigned int)d1, (int)(unsigned int)(d1 >> 32),
                    (int)(unsigned int)d2, (int)(unsigned int)(d2 >> 32),
                    (int)(unsigned int)d3, (int)(unsigned int)(d3 >> 32)};
    }

    if (k0 + 64 < K) {
      __builtin_prefetch(A + (size_t)(mBase + r) * K + k0 + 64, 0, 1);
      __builtin_prefetch(Bw + (size_t)(nBase + r) * K + k0 + 64, 0, 1);
    }

#pragma unroll
    for (int i = 0; i < 4; ++i)
#pragma unroll
      for (int j = 0; j < 2; ++j)
        acc[i][j] = __builtin_amdgcn_wmma_i32_16x16x64_iu8(
             true, aF[i],  true, bF[j], acc[i][j],
             false,  false);
  }

  const float scl = scaleP[0] * (1.0f / 128.0f);
#pragma unroll
  for (int j = 0; j < 2; ++j) {
    const int n = nBase + 16 * j + r;
    const float bn = bias[n];
#pragma unroll
    for (int i = 0; i < 4; ++i) {
      const int mT = mBase + 16 * i + 8 * hi;
#pragma unroll
      for (int vv = 0; vv < 8; ++vv) {
        const int m = mT + vv;
        out[(size_t)m * N + n] = (float)acc[i][j][vv] * (scl * gamma[m]) + bn;
      }
    }
  }
}

extern "C" void kernel_launch(void* const* d_in, const int* in_sizes, int n_in,
                              void* d_out, int out_size, void* d_ws,
                              size_t ws_size, hipStream_t stream) {
  const float* x = (const float*)d_in[0];
  const float* w = (const float*)d_in[1];
  const float* scale = (const float*)d_in[2];
  const float* bias = (const float*)d_in[3];
  float* out = (float*)d_out;

  const int N = in_sizes[3];
  const int K = in_sizes[1] / N;
  const int M = in_sizes[0] / K;

  signed char* ws = (signed char*)d_ws;
  signed char* xq = ws;
  signed char* wq = ws + (size_t)M * K;
  float* gamma = (float*)(ws + (size_t)M * K + (size_t)N * K);

  quant_act_kernel<<<M / 8, 256, 0, stream>>>(x, xq, gamma, M);
  quant_wgt_kernel<<<((N * K / 4) + 255) / 256, 256, 0, stream>>>(w, wq,
                                                                  N * K / 4);
  dim3 grid(M / 128, N / 128);
  bitgemm_kernel<<<grid, 256, 0, stream>>>(xq, wq, gamma, scale, bias, out, M,
                                           N, K);
}
